// EdgeOnlyMPNN_62680752718359
// MI455X (gfx1250) — matrix operand order repaired and checked by reading, not compiled, not run on hardware
//
#include <hip/hip_runtime.h>
#include <hip/hip_bf16.h>

// ---------------- problem constants ----------------
constexpr int N_NODES = 8192;
constexpr int N_EDGES = 131072;
constexpr int BATCH   = 2;
constexpr int HID     = 256;
constexpr int MSG     = 256;
constexpr int EDIM    = 64;
constexpr int MLP_H   = 512;
constexpr int OUT     = 128;
constexpr int N_ITERS = 3;
constexpr int K1      = HID + HID + EDIM;   // 576
constexpr int NPAIRS  = N_NODES * BATCH;    // 16384 node-pair rows
constexpr int EPAIRS  = N_EDGES * BATCH;    // 262144 edge-pair rows
constexpr int EROWS   = 64;                 // edge rows per block (4 subtiles of 16)

typedef _Float16 half8  __attribute__((ext_vector_type(8)));
typedef _Float16 half16 __attribute__((ext_vector_type(16)));
typedef float    float8 __attribute__((ext_vector_type(8)));

#define WMMA16(A, B, C) \
    __builtin_amdgcn_wmma_f32_16x16x32_f16(false, (A), false, (B), (short)0, (C), false, false)

// ---------------- fragment loaders ----------------
// A: 16x32 f16 (M x K).  lanes 0-15: M=lane, K groups {0..7,16..23};
// lanes 16-31: M=lane-16, K groups {8..15,24..31}.
__device__ __forceinline__ half16 ldA(const _Float16* s, int row0, int k0, int ld, int lane) {
    int row = row0 + (lane & 15);
    int kb  = k0 + ((lane >> 4) << 3);
    const _Float16* base = s + row * ld + kb;
    half8 lo = *reinterpret_cast<const half8*>(base);
    half8 hi = *reinterpret_cast<const half8*>(base + 16);
    return __builtin_shufflevector(lo, hi, 0,1,2,3,4,5,6,7,8,9,10,11,12,13,14,15);
}

// B: 32x16 f16 (K x N), column-major weights wcm[col*K + k].
// lanes 0-15: N=lane, K groups {0..7,16..23} ; lanes 16-31: N=lane-16, K groups {8..15,24..31}.
__device__ __forceinline__ half16 ldB(const _Float16* w, int col0, int k0, int K, int lane) {
    int col = col0 + (lane & 15);
    int kb  = k0 + ((lane >> 4) << 3);
    const half8* p = reinterpret_cast<const half8*>(w + col * K + kb);
    half8 lo = p[0];
    half8 hi = p[2];
    return __builtin_shufflevector(lo, hi, 0,1,2,3,4,5,6,7,8,9,10,11,12,13,14,15);
}

__device__ __forceinline__ float sigmoidf_(float x) {
    return 1.0f / (1.0f + __expf(-x));
}

// ---------------- prep kernels ----------------
__global__ void zero_f32_kernel(float* __restrict__ p, int n) {
    int i = blockIdx.x * blockDim.x + threadIdx.x;
    if (i < n) p[i] = 0.0f;
}

// dst[n*K + k] = (f16) src[k*N + n]   (row-major (K,N) -> col-major f16)
__global__ void convert_cm_kernel(const float* __restrict__ src, _Float16* __restrict__ dst,
                                  int K, int N) {
    int i = blockIdx.x * blockDim.x + threadIdx.x;
    if (i < K * N) {
        int n = i / K;
        int k = i - n * K;
        dst[n * K + k] = (_Float16)src[k * N + n];
    }
}

// ---------------- edge message MLP + scatter ----------------
// One block: 64 (edge,batch) rows. 256 threads = 8 waves.
// Dynamic LDS: 64*576 f16 (xs), reused as 64*512 f16 (ys).
__global__ __launch_bounds__(256) void edge_mlp_kernel(
    const float* __restrict__ h, const float* __restrict__ edge_feat,
    const int* __restrict__ edge_src, const int* __restrict__ edge_dst,
    const _Float16* __restrict__ w1cm, const float* __restrict__ b1,
    const _Float16* __restrict__ w2cm, const float* __restrict__ b2,
    float* __restrict__ m)
{
    extern __shared__ __align__(16) char edge_smem_raw[];
    _Float16* smem = reinterpret_cast<_Float16*>(edge_smem_raw);
    __shared__ int dstrow[EROWS];
    __shared__ int srcrow[EROWS];

    const int tid  = threadIdx.x;
    const int lane = tid & 31;
    const int wv   = tid >> 5;
    const int p0   = blockIdx.x * EROWS;

    if (tid < EROWS) {
        int e = (p0 + tid) >> 1;
        dstrow[tid] = edge_dst[e];
        srcrow[tid] = edge_src[e];
    }
    __syncthreads();

    // ---- stage 1: gather x = [h_dst | h_src | edge_feat] as f16 into LDS ----
    for (int i = tid; i < EROWS * K1; i += 256) {
        int r = i / K1;
        int c = i - r * K1;
        int p = p0 + r;
        int b = p & 1;
        float val;
        if (c < HID) {
            val = h[(dstrow[r] * 2 + b) * HID + c];
        } else if (c < 2 * HID) {
            val = h[(srcrow[r] * 2 + b) * HID + (c - HID)];
        } else {
            val = edge_feat[((p >> 1) * 2 + b) * EDIM + (c - 2 * HID)];
        }
        smem[r * K1 + c] = (_Float16)val;
    }
    __syncthreads();

    // ---- stage 2: layer1  (64 x 576) @ (576 x 512), wave owns 64 cols ----
    float8 acc1[4][4] = {};
    const int n1 = wv * 64;
    for (int kk = 0; kk < K1; kk += 32) {
        half16 a0 = ldA(smem, 0,  kk, K1, lane);
        half16 a1 = ldA(smem, 16, kk, K1, lane);
        half16 a2 = ldA(smem, 32, kk, K1, lane);
        half16 a3 = ldA(smem, 48, kk, K1, lane);
#pragma unroll
        for (int t = 0; t < 4; ++t) {
            // prefetch next k-step's fragment for this lane (gfx1250 global_prefetch)
            if (kk + 32 < K1) {
                int col = n1 + t * 16 + (lane & 15);
                int kb  = kk + 32 + ((lane >> 4) << 4);
                __builtin_prefetch(w1cm + col * K1 + kb, 0, 3);
            }
            half16 bf = ldB(w1cm, n1 + t * 16, kk, K1, lane);
            acc1[0][t] = WMMA16(a0, bf, acc1[0][t]);
            acc1[1][t] = WMMA16(a1, bf, acc1[1][t]);
            acc1[2][t] = WMMA16(a2, bf, acc1[2][t]);
            acc1[3][t] = WMMA16(a3, bf, acc1[3][t]);
        }
    }
    __syncthreads();   // everyone done reading xs before overwriting as ys

    // bias + relu, store f16 activations (stride 512) into same LDS
#pragma unroll
    for (int s = 0; s < 4; ++s)
#pragma unroll
        for (int t = 0; t < 4; ++t)
#pragma unroll
            for (int v = 0; v < 8; ++v) {
                int row = s * 16 + ((lane >> 4) << 3) + v;
                int col = n1 + t * 16 + (lane & 15);
                float x = acc1[s][t][v] + b1[col];
                smem[row * MLP_H + col] = (_Float16)(x > 0.0f ? x : 0.0f);
            }
    __syncthreads();

    // ---- stage 3: layer2  (64 x 512) @ (512 x 256), wave owns 32 cols ----
    float8 acc2[4][2] = {};
    const int n2 = wv * 32;
    for (int kk = 0; kk < MLP_H; kk += 32) {
        half16 a0 = ldA(smem, 0,  kk, MLP_H, lane);
        half16 a1 = ldA(smem, 16, kk, MLP_H, lane);
        half16 a2 = ldA(smem, 32, kk, MLP_H, lane);
        half16 a3 = ldA(smem, 48, kk, MLP_H, lane);
#pragma unroll
        for (int t = 0; t < 2; ++t) {
            half16 bf = ldB(w2cm, n2 + t * 16, kk, MLP_H, lane);
            acc2[0][t] = WMMA16(a0, bf, acc2[0][t]);
            acc2[1][t] = WMMA16(a1, bf, acc2[1][t]);
            acc2[2][t] = WMMA16(a2, bf, acc2[2][t]);
            acc2[3][t] = WMMA16(a3, bf, acc2[3][t]);
        }
    }

    // ---- stage 4: msg + b2, atomic scatter-add into m[dst] ----
#pragma unroll
    for (int s = 0; s < 4; ++s)
#pragma unroll
        for (int t = 0; t < 2; ++t)
#pragma unroll
            for (int v = 0; v < 8; ++v) {
                int row = s * 16 + ((lane >> 4) << 3) + v;
                int col = n2 + t * 16 + (lane & 15);
                int p   = p0 + row;
                int b   = p & 1;
                float val = acc2[s][t][v] + b2[col];
                atomicAdd(&m[(dstrow[row] * 2 + b) * MSG + col], val);
            }
}

// ---------------- GRU vertex update ----------------
// One block: 32 node-pair rows. r/z/h~ via WMMA, elementwise in VALU.
__global__ __launch_bounds__(256) void gru_kernel(
    float* __restrict__ h, const float* __restrict__ m,
    const _Float16* __restrict__ wrcm, const _Float16* __restrict__ urcm, const float* __restrict__ br,
    const _Float16* __restrict__ wzcm, const _Float16* __restrict__ uzcm, const float* __restrict__ bz,
    const _Float16* __restrict__ whcm, const _Float16* __restrict__ uhcm, const float* __restrict__ bh)
{
    __shared__ __align__(16) _Float16 ms16[32 * HID];
    __shared__ __align__(16) _Float16 hs16[32 * HID];
    __shared__ __align__(16) _Float16 rh16[32 * HID];

    const int tid  = threadIdx.x;
    const int lane = tid & 31;
    const int wv   = tid >> 5;
    const int p0   = blockIdx.x * 32;

    for (int i = tid; i < 32 * HID; i += 256) {
        int r = i >> 8;
        int c = i & 255;
        ms16[i] = (_Float16)m[(p0 + r) * HID + c];
        hs16[i] = (_Float16)h[(p0 + r) * HID + c];
    }
    __syncthreads();

    const int nb = wv * 32;   // wave owns 32 output cols

    // ---- r = sigmoid(m@Wr + h@Ur + br);  rh = r * h ----
    float8 accr[2][2] = {};
    for (int kk = 0; kk < HID; kk += 32) {
        half16 am0 = ldA(ms16, 0,  kk, HID, lane);
        half16 am1 = ldA(ms16, 16, kk, HID, lane);
        half16 ah0 = ldA(hs16, 0,  kk, HID, lane);
        half16 ah1 = ldA(hs16, 16, kk, HID, lane);
#pragma unroll
        for (int t = 0; t < 2; ++t) {
            half16 bw = ldB(wrcm, nb + t * 16, kk, HID, lane);
            half16 bu = ldB(urcm, nb + t * 16, kk, HID, lane);
            accr[0][t] = WMMA16(am0, bw, accr[0][t]);
            accr[0][t] = WMMA16(ah0, bu, accr[0][t]);
            accr[1][t] = WMMA16(am1, bw, accr[1][t]);
            accr[1][t] = WMMA16(ah1, bu, accr[1][t]);
        }
    }
#pragma unroll
    for (int s = 0; s < 2; ++s)
#pragma unroll
        for (int t = 0; t < 2; ++t)
#pragma unroll
            for (int v = 0; v < 8; ++v) {
                int row = s * 16 + ((lane >> 4) << 3) + v;
                int col = nb + t * 16 + (lane & 15);
                float rr = sigmoidf_(accr[s][t][v] + br[col]);
                float hv = h[(p0 + row) * HID + col];
                rh16[row * HID + col] = (_Float16)(rr * hv);
            }
    __syncthreads();

    // ---- z = sigmoid(m@Wz + h@Uz + bz);  h~ = tanh(m@Wh + rh@Uh + bh) (fused K loop) ----
    float8 accz[2][2] = {};
    float8 acch[2][2] = {};
    for (int kk = 0; kk < HID; kk += 32) {
        half16 am0 = ldA(ms16, 0,  kk, HID, lane);
        half16 am1 = ldA(ms16, 16, kk, HID, lane);
        half16 ah0 = ldA(hs16, 0,  kk, HID, lane);
        half16 ah1 = ldA(hs16, 16, kk, HID, lane);
        half16 ar0 = ldA(rh16, 0,  kk, HID, lane);
        half16 ar1 = ldA(rh16, 16, kk, HID, lane);
#pragma unroll
        for (int t = 0; t < 2; ++t) {
            half16 bwz = ldB(wzcm, nb + t * 16, kk, HID, lane);
            half16 buz = ldB(uzcm, nb + t * 16, kk, HID, lane);
            half16 bwh = ldB(whcm, nb + t * 16, kk, HID, lane);
            half16 buh = ldB(uhcm, nb + t * 16, kk, HID, lane);
            accz[0][t] = WMMA16(am0, bwz, accz[0][t]);
            accz[0][t] = WMMA16(ah0, buz, accz[0][t]);
            accz[1][t] = WMMA16(am1, bwz, accz[1][t]);
            accz[1][t] = WMMA16(ah1, buz, accz[1][t]);
            acch[0][t] = WMMA16(am0, bwh, acch[0][t]);
            acch[0][t] = WMMA16(ar0, buh, acch[0][t]);
            acch[1][t] = WMMA16(am1, bwh, acch[1][t]);
            acch[1][t] = WMMA16(ar1, buh, acch[1][t]);
        }
    }

    // ---- h = (1-z)*h + z*h~ ----
#pragma unroll
    for (int s = 0; s < 2; ++s)
#pragma unroll
        for (int t = 0; t < 2; ++t)
#pragma unroll
            for (int v = 0; v < 8; ++v) {
                int row = s * 16 + ((lane >> 4) << 3) + v;
                int col = nb + t * 16 + (lane & 15);
                float z  = sigmoidf_(accz[s][t][v] + bz[col]);
                float ht = tanhf(acch[s][t][v] + bh[col]);
                float hv = h[(p0 + row) * HID + col];
                h[(p0 + row) * HID + col] = (1.0f - z) * hv + z * ht;
            }
}

// ---------------- readout ----------------
// phase 1: partial node sums, 64 blocks x 128 nodes, atomic into g[2*256]
__global__ __launch_bounds__(256) void readout_sum_kernel(
    const float* __restrict__ h, float* __restrict__ g)
{
    const int tid = threadIdx.x;
    const int n0  = blockIdx.x * 128;
    for (int i = tid; i < BATCH * HID; i += 256) {
        int b = i >> 8;
        int c = i & 255;
        float s = 0.0f;
        for (int n = n0; n < n0 + 128; ++n)
            s += h[(n * 2 + b) * HID + c];
        atomicAdd(&g[i], s);
    }
}

// phase 2: out = g @ Wo + bo   (2x256 @ 256x128)
__global__ __launch_bounds__(256) void readout_out_kernel(
    const float* __restrict__ g, const float* __restrict__ Wo,
    const float* __restrict__ bo, float* __restrict__ out)
{
    const int tid = threadIdx.x;
    int b = tid >> 7;
    int o = tid & 127;
    float acc = bo[o];
    for (int k = 0; k < HID; ++k)
        acc += g[b * HID + k] * Wo[k * OUT + o];
    out[b * OUT + o] = acc;
}

// ---------------- host launch ----------------
extern "C" void kernel_launch(void* const* d_in, const int* in_sizes, int n_in,
                              void* d_out, int out_size, void* d_ws, size_t ws_size,
                              hipStream_t stream) {
    const float* edge_feat = (const float*)d_in[0];
    const float* W1 = (const float*)d_in[1];
    const float* b1 = (const float*)d_in[2];
    const float* W2 = (const float*)d_in[3];
    const float* b2 = (const float*)d_in[4];
    const float* Wr = (const float*)d_in[5];
    const float* Ur = (const float*)d_in[6];
    const float* br = (const float*)d_in[7];
    const float* Wz = (const float*)d_in[8];
    const float* Uz = (const float*)d_in[9];
    const float* bz = (const float*)d_in[10];
    const float* Wh = (const float*)d_in[11];
    const float* Uh = (const float*)d_in[12];
    const float* bh = (const float*)d_in[13];
    const float* Wo = (const float*)d_in[14];
    const float* bo = (const float*)d_in[15];
    const int* edge_src = (const int*)d_in[16];
    const int* edge_dst = (const int*)d_in[17];
    float* out = (float*)d_out;

    // workspace carve-up (256B aligned)
    char* ws = (char*)d_ws;
    size_t off = 0;
    auto carve = [&](size_t bytes) {
        void* p = ws + off;
        off += (bytes + 255) & ~(size_t)255;
        return p;
    };
    float*     h    = (float*)    carve((size_t)NPAIRS * HID * 4);     // 16 MB
    float*     m    = (float*)    carve((size_t)NPAIRS * MSG * 4);     // 16 MB
    _Float16*  w1cm = (_Float16*) carve((size_t)K1 * MLP_H * 2);
    _Float16*  w2cm = (_Float16*) carve((size_t)MLP_H * MSG * 2);
    _Float16*  wrcm = (_Float16*) carve((size_t)HID * HID * 2);
    _Float16*  urcm = (_Float16*) carve((size_t)HID * HID * 2);
    _Float16*  wzcm = (_Float16*) carve((size_t)HID * HID * 2);
    _Float16*  uzcm = (_Float16*) carve((size_t)HID * HID * 2);
    _Float16*  whcm = (_Float16*) carve((size_t)HID * HID * 2);
    _Float16*  uhcm = (_Float16*) carve((size_t)HID * HID * 2);
    float*     g    = (float*)    carve((size_t)BATCH * HID * 4);
    (void)ws_size; (void)in_sizes; (void)n_in; (void)out_size;

    const int T = 256;
    // zero node state (h, m persist/accumulate across the 3 iterations) and g
    zero_f32_kernel<<<(NPAIRS * HID + T - 1) / T, T, 0, stream>>>(h, NPAIRS * HID);
    zero_f32_kernel<<<(NPAIRS * MSG + T - 1) / T, T, 0, stream>>>(m, NPAIRS * MSG);
    zero_f32_kernel<<<(BATCH * HID + T - 1) / T, T, 0, stream>>>(g, BATCH * HID);

    // convert weights to f16 column-major
    convert_cm_kernel<<<(K1 * MLP_H + T - 1) / T, T, 0, stream>>>(W1, w1cm, K1, MLP_H);
    convert_cm_kernel<<<(MLP_H * MSG + T - 1) / T, T, 0, stream>>>(W2, w2cm, MLP_H, MSG);
    int nsq = HID * HID;
    convert_cm_kernel<<<(nsq + T - 1) / T, T, 0, stream>>>(Wr, wrcm, HID, HID);
    convert_cm_kernel<<<(nsq + T - 1) / T, T, 0, stream>>>(Ur, urcm, HID, HID);
    convert_cm_kernel<<<(nsq + T - 1) / T, T, 0, stream>>>(Wz, wzcm, HID, HID);
    convert_cm_kernel<<<(nsq + T - 1) / T, T, 0, stream>>>(Uz, uzcm, HID, HID);
    convert_cm_kernel<<<(nsq + T - 1) / T, T, 0, stream>>>(Wh, whcm, HID, HID);
    convert_cm_kernel<<<(nsq + T - 1) / T, T, 0, stream>>>(Uh, uhcm, HID, HID);

    const size_t edge_lds = (size_t)EROWS * K1 * sizeof(_Float16);   // 73,728 B
    for (int it = 0; it < N_ITERS; ++it) {
        edge_mlp_kernel<<<EPAIRS / EROWS, T, edge_lds, stream>>>(
            h, edge_feat, edge_src, edge_dst, w1cm, b1, w2cm, b2, m);
        gru_kernel<<<NPAIRS / 32, T, 0, stream>>>(
            h, m, wrcm, urcm, br, wzcm, uzcm, bz, whcm, uhcm, bh);
    }

    readout_sum_kernel<<<N_NODES / 128, T, 0, stream>>>(h, g);
    readout_out_kernel<<<1, T, 0, stream>>>(g, Wo, bo, out);
}
